// meta_optimizer_68015102099898
// MI455X (gfx1250) — matrix operand order repaired and checked by reading, not compiled, not run on hardware
//
#include <hip/hip_runtime.h>
#include <hip/hip_bf16.h>
#include <math.h>

// Problem dims (fixed by the reference)
#define NB 128   // batch
#define RD 2304  // rows (tall)
#define CD 256   // cols

typedef __attribute__((ext_vector_type(16))) __bf16 v16bf;
typedef __attribute__((ext_vector_type(8)))  float  v8f;

// ---------------------------------------------------------------------------
// fp32 -> bf16 hi/lo split (for 3-term split-precision WMMA accumulation)
// ---------------------------------------------------------------------------
__device__ __forceinline__ void split_bf16(float x, __bf16 &hi, __bf16 &lo) {
  const __bf16 h = (__bf16)x;
  hi = h;
  lo = (__bf16)(x - (float)h);
}

// ---------------------------------------------------------------------------
// Fragment loaders matching the CDNA5 16-bit WMMA VGPR layouts (ISA 7.12.2).
// A (16x32, MxK): lanes 0-15 -> M=lane, K-base 0; lanes 16-31 -> M=lane-16,
//   K-base 8; element e (v=e>>1,pos=e&1): K = 16*(v>>2) + kbase + 2*(v&3)+pos.
// B (32x16, KxN): lanes 0-15 -> N=lane, K-base 0; lanes 16-31 -> N=lane-16,
//   K-base 8; element e: K = kbase + e (e<8), 16 + kbase + (e-8) (e>=8).
// src is an LDS fp32 tile addressed as src[m*sm + k*sk].
// ---------------------------------------------------------------------------
__device__ __forceinline__ void load_frag_a(const float* src, int sm, int sk,
                                            int row_base, int lane,
                                            v16bf &hi, v16bf &lo) {
  const int m = row_base + (lane & 15);
  const int kbase = (lane < 16) ? 0 : 8;
#pragma unroll
  for (int e = 0; e < 16; ++e) {
    const int v = e >> 1, pos = e & 1;
    const int kk = ((v >> 2) << 4) + kbase + ((v & 3) << 1) + pos;
    __bf16 h, l;
    split_bf16(src[m * sm + kk * sk], h, l);
    hi[e] = h; lo[e] = l;
  }
}

__device__ __forceinline__ void load_frag_b(const float* src, int sk, int sn,
                                            int col_base, int lane,
                                            v16bf &hi, v16bf &lo) {
  const int n = col_base + (lane & 15);
  const int kbase = (lane < 16) ? 0 : 8;
#pragma unroll
  for (int e = 0; e < 16; ++e) {
    __bf16 h, l;
    split_bf16(src[(kbase + e + ((e >> 3) << 3)) * sk + n * sn], h, l);
    hi[e] = h; lo[e] = l;
  }
}

// acc += A*B in ~fp32 precision via 3 bf16 WMMAs (hi*hi + hi*lo + lo*hi)
__device__ __forceinline__ v8f wmma_split(v16bf ah, v16bf al,
                                          v16bf bh, v16bf bl, v8f acc) {
  acc = __builtin_amdgcn_wmma_f32_16x16x32_bf16(false, al, false, bh, (short)0, acc, false, false);
  acc = __builtin_amdgcn_wmma_f32_16x16x32_bf16(false, ah, false, bl, (short)0, acc, false, false);
  acc = __builtin_amdgcn_wmma_f32_16x16x32_bf16(false, ah, false, bh, (short)0, acc, false, false);
  return acc;
}

// ---------------------------------------------------------------------------
// Kernel 1/4: Out[n] = X[n]^T * Y[n]   (X,Y: [N][K][256], Out: [N][256][256])
// 128x64 output tile per workgroup; 8 wave32s, each owning a 2x2 block of
// 16x16 accumulators (12 WMMAs per K-step per wave).
// ---------------------------------------------------------------------------
__global__ void __launch_bounds__(256)
gram_kernel(const float* X, const float* Y, float* __restrict__ Out, int K) {
  const int n = blockIdx.z;
  const int i0 = blockIdx.x * 128, j0 = blockIdx.y * 64;
  const float* Xn = X + (size_t)n * K * CD;
  const float* Yn = Y + (size_t)n * K * CD;
  __shared__ float Xs[32][132];  // [k][i], padded
  __shared__ float Ys[32][68];   // [k][j], padded
  const int t = threadIdx.x, lane = t & 31, wid = t >> 5;
  const int ti0 = (wid & 3) * 2, tj0 = (wid >> 2) * 2;
  v8f acc[2][2] = {};
  for (int k0 = 0; k0 < K; k0 += 32) {
    __syncthreads();
#pragma unroll
    for (int l = 0; l < 16; ++l) {          // X tile: 32x128
      const int idx = t + l * 256;
      const int kk = idx >> 7, cc = idx & 127;
      Xs[kk][cc] = Xn[(size_t)(k0 + kk) * CD + i0 + cc];
    }
#pragma unroll
    for (int l = 0; l < 8; ++l) {           // Y tile: 32x64
      const int idx = t + l * 256;
      const int kk = idx >> 6, cc = idx & 63;
      Ys[kk][cc] = Yn[(size_t)(k0 + kk) * CD + j0 + cc];
    }
    // stream next K-tile toward the caches while this one is consumed
    if (k0 + 32 < K) {
      const int pk = t >> 3, pc = (t & 7) * 16;
      __builtin_prefetch(&Xn[(size_t)(k0 + 32 + pk) * CD + i0 + pc], 0, 0);
      if (t < 128) {
        const int qk = t >> 2, qc = (t & 3) * 16;
        __builtin_prefetch(&Yn[(size_t)(k0 + 32 + qk) * CD + j0 + qc], 0, 0);
      }
    }
    __syncthreads();
    v16bf ah[2], al[2], bh[2], bl[2];
#pragma unroll
    for (int ii = 0; ii < 2; ++ii)   // A = X^T: A_op[i][k] = Xs[k][i]
      load_frag_a(&Xs[0][0], /*sm=*/1, /*sk=*/132, (ti0 + ii) * 16, lane, ah[ii], al[ii]);
#pragma unroll
    for (int jj = 0; jj < 2; ++jj)
      load_frag_b(&Ys[0][0], /*sk=*/68, /*sn=*/1, (tj0 + jj) * 16, lane, bh[jj], bl[jj]);
#pragma unroll
    for (int ii = 0; ii < 2; ++ii)
#pragma unroll
      for (int jj = 0; jj < 2; ++jj)
        acc[ii][jj] = wmma_split(ah[ii], al[ii], bh[jj], bl[jj], acc[ii][jj]);
  }
  float* On = Out + (size_t)n * CD * CD;
  const int nn = lane & 15, mofs = (lane < 16) ? 0 : 8;
#pragma unroll
  for (int ii = 0; ii < 2; ++ii)
#pragma unroll
    for (int jj = 0; jj < 2; ++jj)
#pragma unroll
      for (int v = 0; v < 8; ++v) {
        const int m = i0 + (ti0 + ii) * 16 + mofs + v;  // C/D layout: VGPR v -> row
        On[(size_t)m * CD + j0 + (tj0 + jj) * 16 + nn] = acc[ii][jj][v];
      }
}

// ---------------------------------------------------------------------------
// Kernel 2: in-place A <- 0.5*(A + A^T), per batch
// ---------------------------------------------------------------------------
__global__ void __launch_bounds__(256)
symmetrize_kernel(float* __restrict__ A) {
  const int n = blockIdx.y;
  const int idx = blockIdx.x * 256 + threadIdx.x;  // over 256*256
  const int i = idx >> 8, j = idx & 255;
  if (i < j) {
    float* An = A + (size_t)n * CD * CD;
    const float a = An[i * CD + j], b = An[j * CD + i];
    const float s = 0.5f * (a + b);
    An[i * CD + j] = s;
    An[j * CD + i] = s;
  }
}

// ---------------------------------------------------------------------------
// Kernel 3: W = M - a*(U + G) + a*(M @ Asym),  a = |lr[n]| * s_lr
// (since G_proj = G - M@Asym and W = M - a*(U + G_proj))
// 128x64 output tile per workgroup, 2x2 accumulator block per wave.
// ---------------------------------------------------------------------------
__global__ void __launch_bounds__(256)
build_w_kernel(const float* __restrict__ M, const float* __restrict__ G,
               const float* __restrict__ U, const float* __restrict__ lr,
               const float* __restrict__ s_lr, const float* __restrict__ Asym,
               float* __restrict__ W) {
  const int n = blockIdx.z;
  const int r0 = blockIdx.x * 128, c0 = blockIdx.y * 64;
  const float a = fabsf(lr[n]) * s_lr[0];
  const float* Mn = M + (size_t)n * RD * CD;
  const float* Gn = G + (size_t)n * RD * CD;
  const float* Un = U + (size_t)n * RD * CD;
  const float* An = Asym + (size_t)n * CD * CD;
  float* Wn = W + (size_t)n * RD * CD;
  __shared__ float Ms[128][36];  // [m][k], padded
  __shared__ float Bs[32][68];   // [k][c], padded
  const int t = threadIdx.x, lane = t & 31, wid = t >> 5;
  const int ti0 = (wid & 3) * 2, tj0 = (wid >> 2) * 2;
  v8f acc[2][2] = {};
  for (int k0 = 0; k0 < CD; k0 += 32) {
    __syncthreads();
#pragma unroll
    for (int l = 0; l < 16; ++l) {          // M tile: 128x32
      const int idx = t + l * 256;
      const int m = idx >> 5, kk = idx & 31;
      Ms[m][kk] = Mn[(size_t)(r0 + m) * CD + k0 + kk];
    }
#pragma unroll
    for (int l = 0; l < 8; ++l) {           // Asym tile: 32x64
      const int idx = t + l * 256;
      const int kk = idx >> 6, cc = idx & 63;
      Bs[kk][cc] = An[(size_t)(k0 + kk) * CD + c0 + cc];
    }
    if (k0 + 32 < CD) {
      if (t < 128) __builtin_prefetch(&Mn[(size_t)(r0 + t) * CD + k0 + 32], 0, 0);
      if (t < 64) {
        const int qk = t >> 1, qc = (t & 1) * 32;
        __builtin_prefetch(&An[(size_t)(k0 + 32 + qk) * CD + c0 + qc], 0, 0);
      }
    }
    __syncthreads();
    v16bf ah[2], al[2], bh[2], bl[2];
#pragma unroll
    for (int ii = 0; ii < 2; ++ii)
      load_frag_a(&Ms[0][0], /*sm=*/36, /*sk=*/1, (ti0 + ii) * 16, lane, ah[ii], al[ii]);
#pragma unroll
    for (int jj = 0; jj < 2; ++jj)
      load_frag_b(&Bs[0][0], 68, 1, (tj0 + jj) * 16, lane, bh[jj], bl[jj]);
#pragma unroll
    for (int ii = 0; ii < 2; ++ii)
#pragma unroll
      for (int jj = 0; jj < 2; ++jj)
        acc[ii][jj] = wmma_split(ah[ii], al[ii], bh[jj], bl[jj], acc[ii][jj]);
  }
  const int nn = lane & 15, mofs = (lane < 16) ? 0 : 8;
#pragma unroll
  for (int ii = 0; ii < 2; ++ii)
#pragma unroll
    for (int jj = 0; jj < 2; ++jj)
#pragma unroll
      for (int v = 0; v < 8; ++v) {
        const int m = r0 + (ti0 + ii) * 16 + mofs + v;
        const size_t off = (size_t)m * CD + c0 + (tj0 + jj) * 16 + nn;
        Wn[off] = Mn[off] - a * (Un[off] + Gn[off]) + a * acc[ii][jj][v];
      }
}

// ---------------------------------------------------------------------------
// Kernel 5: per-batch Cholesky of S (upper, in place: S -> R with R^T R = S,
// diag(R) > 0) followed by T = R^{-1} (upper-triangular inverse).
// One workgroup (256 threads) per batch item. Positive diag of R means
// Q = W * T is exactly the sign-fixed Q of the reference.
// ---------------------------------------------------------------------------
__global__ void __launch_bounds__(256)
chol_inv_kernel(float* __restrict__ S, float* __restrict__ T) {
  const int n = blockIdx.x, t = threadIdx.x;
  float* A  = S + (size_t)n * CD * CD;
  float* Tn = T + (size_t)n * CD * CD;
  for (int k = 0; k < CD; ++k) {
    if (t == 0) A[k * CD + k] = sqrtf(A[k * CD + k]);
    __syncthreads();
    const float rkk = A[k * CD + k];
    const int j = k + 1 + t;
    if (j < CD) A[k * CD + j] /= rkk;
    __syncthreads();
    for (int i = k + 1; i < CD; ++i) {          // rank-1 update, upper part
      const float rki = A[k * CD + i];
      const int jj = i + t;
      if (jj < CD) A[i * CD + jj] -= rki * A[k * CD + jj];
    }
    __syncthreads();
  }
  // Triangular inverse: thread t owns column j = t of T (back substitution).
  const int j = t;
  for (int i = j + 1; i < CD; ++i) Tn[i * CD + j] = 0.0f;  // lower zeros
  Tn[j * CD + j] = 1.0f / A[j * CD + j];
  for (int i = j - 1; i >= 0; --i) {
    float s = 0.0f;
    for (int k = i + 1; k <= j; ++k) s += A[i * CD + k] * Tn[k * CD + j];
    Tn[i * CD + j] = -s / A[i * CD + i];
  }
}

// ---------------------------------------------------------------------------
// Kernel 6: Out[n][c][r] = (W @ T)[r][c]   (transposed store epilogue)
// ---------------------------------------------------------------------------
__global__ void __launch_bounds__(256)
qt_kernel(const float* __restrict__ W, const float* __restrict__ T,
          float* __restrict__ Out) {
  const int n = blockIdx.z;
  const int r0 = blockIdx.x * 128, c0 = blockIdx.y * 64;
  const float* Wn = W + (size_t)n * RD * CD;
  const float* Tn = T + (size_t)n * CD * CD;
  float* On = Out + (size_t)n * CD * RD;
  __shared__ float Ms[128][36];
  __shared__ float Bs[32][68];
  const int t = threadIdx.x, lane = t & 31, wid = t >> 5;
  const int ti0 = (wid & 3) * 2, tj0 = (wid >> 2) * 2;
  v8f acc[2][2] = {};
  for (int k0 = 0; k0 < CD; k0 += 32) {
    __syncthreads();
#pragma unroll
    for (int l = 0; l < 16; ++l) {
      const int idx = t + l * 256;
      const int m = idx >> 5, kk = idx & 31;
      Ms[m][kk] = Wn[(size_t)(r0 + m) * CD + k0 + kk];
    }
#pragma unroll
    for (int l = 0; l < 8; ++l) {
      const int idx = t + l * 256;
      const int kk = idx >> 6, cc = idx & 63;
      Bs[kk][cc] = Tn[(size_t)(k0 + kk) * CD + c0 + cc];
    }
    if (k0 + 32 < CD) {
      if (t < 128) __builtin_prefetch(&Wn[(size_t)(r0 + t) * CD + k0 + 32], 0, 0);
      if (t < 64) {
        const int qk = t >> 1, qc = (t & 1) * 32;
        __builtin_prefetch(&Tn[(size_t)(k0 + 32 + qk) * CD + c0 + qc], 0, 0);
      }
    }
    __syncthreads();
    v16bf ah[2], al[2], bh[2], bl[2];
#pragma unroll
    for (int ii = 0; ii < 2; ++ii)
      load_frag_a(&Ms[0][0], 36, 1, (ti0 + ii) * 16, lane, ah[ii], al[ii]);
#pragma unroll
    for (int jj = 0; jj < 2; ++jj)
      load_frag_b(&Bs[0][0], 68, 1, (tj0 + jj) * 16, lane, bh[jj], bl[jj]);
#pragma unroll
    for (int ii = 0; ii < 2; ++ii)
#pragma unroll
      for (int jj = 0; jj < 2; ++jj)
        acc[ii][jj] = wmma_split(ah[ii], al[ii], bh[jj], bl[jj], acc[ii][jj]);
  }
  const int nn = lane & 15, mofs = (lane < 16) ? 0 : 8;
#pragma unroll
  for (int ii = 0; ii < 2; ++ii)
#pragma unroll
    for (int jj = 0; jj < 2; ++jj)
#pragma unroll
      for (int v = 0; v < 8; ++v) {
        const int m = r0 + (ti0 + ii) * 16 + mofs + v;
        On[(size_t)(c0 + (tj0 + jj) * 16 + nn) * RD + m] = acc[ii][jj][v];
      }
}

// ---------------------------------------------------------------------------
extern "C" void kernel_launch(void* const* d_in, const int* in_sizes, int n_in,
                              void* d_out, int out_size, void* d_ws, size_t ws_size,
                              hipStream_t stream) {
  (void)in_sizes; (void)n_in; (void)out_size; (void)ws_size;
  const float* M   = (const float*)d_in[0];  // [N][R][C]
  const float* G   = (const float*)d_in[1];  // [N][R][C]
  const float* U   = (const float*)d_in[2];  // [N][R][C]
  const float* lr  = (const float*)d_in[3];  // [N]
  const float* slr = (const float*)d_in[4];  // [1]
  float* out = (float*)d_out;                // [N][C][R]

  float* ws  = (float*)d_ws;
  float* wsA = ws;                              // [N][C][C]  A / A_sym
  float* wsS = wsA + (size_t)NB * CD * CD;      // [N][C][C]  S -> R (chol)
  float* wsT = wsS + (size_t)NB * CD * CD;      // [N][C][C]  T = R^{-1}
  float* wsW = wsT + (size_t)NB * CD * CD;      // [N][R][C]  W = M + P

  const dim3 blk(256);
  // 1) A = M^T G
  gram_kernel<<<dim3(CD / 128, CD / 64, NB), blk, 0, stream>>>(M, G, wsA, RD);
  // 2) A <- sym(A)
  symmetrize_kernel<<<dim3((CD * CD) / 256, NB), blk, 0, stream>>>(wsA);
  // 3) W = M - a*(U+G) + a*M@Asym
  build_w_kernel<<<dim3(RD / 128, CD / 64, NB), blk, 0, stream>>>(M, G, U, lr, slr, wsA, wsW);
  // 4) S = W^T W
  gram_kernel<<<dim3(CD / 128, CD / 64, NB), blk, 0, stream>>>(wsW, wsW, wsS, RD);
  // 5) S -> R (Cholesky, positive diag), T = R^{-1}
  chol_inv_kernel<<<dim3(NB), blk, 0, stream>>>(wsS, wsT);
  // 6) Out = (W T)^T  == reference's (Q * sgn)^T
  qt_kernel<<<dim3(RD / 128, CD / 64, NB), blk, 0, stream>>>(wsW, wsT, out);
}
